// sequence_2_sequence_LSTM_6605659701489
// MI455X (gfx1250) — matrix operand order repaired and checked by reading, not compiled, not run on hardware
//
#include <hip/hip_runtime.h>

typedef __attribute__((ext_vector_type(16))) _Float16 v16h;
typedef __attribute__((ext_vector_type(8)))  _Float16 v8h;
typedef __attribute__((ext_vector_type(8)))  float    v8f;

// Problem dimensions (from reference)
constexpr int kB     = 64;
constexpr int kTEnc  = 32;
constexpr int kDIn   = 4096;
constexpr int kTDec  = 20;
constexpr int kDWord = 300;
constexpr int kDWpad = 320;   // pad 300 -> 320 (10 K-tiles of 32)
constexpr int kH     = 512;
constexpr int k4H    = 2048;
constexpr int kNPpad = 320;   // proj N pad 300 -> 320 (20 N-tiles, %4==0)

// ---------------------------------------------------------------------------
// Fragment loaders (CDNA5 ISA 7.12.2 layouts, wave32)
// A (16x32 f16, row-major source): lanes 0-15 row M=lane hold K 0-7 (v0-3) and
// K 16-23 (v4-7); lanes 16-31 row M=lane-16 hold K 8-15 and K 24-31.
__device__ __forceinline__ v16h load_a_frag(const _Float16* __restrict__ A,
                                            int lda, int mt, int kt, int lane) {
  const int m    = mt * 16 + (lane & 15);
  const int half = lane >> 4;
  const _Float16* p = A + (size_t)m * lda + kt * 32 + half * 8;
  v8h lo = *(const v8h*)p;          // K offsets half*8 .. half*8+7
  v8h hi = *(const v8h*)(p + 16);   // K offsets 16+half*8 ..
  v16h r;
#pragma unroll
  for (int i = 0; i < 8; ++i) { r[i] = lo[i]; r[i + 8] = hi[i]; }
  return r;
}

// B (32x16 f16) is pre-packed so each lane reads 16 contiguous f16 (32B).
__device__ __forceinline__ v16h load_b_frag(const _Float16* __restrict__ Bp,
                                            int ntiles, int kt, int nt, int lane) {
  const _Float16* p = Bp + ((size_t)(kt * ntiles + nt) * 512) + lane * 16;
  v8h lo = *(const v8h*)p;
  v8h hi = *(const v8h*)(p + 8);
  v16h r;
#pragma unroll
  for (int i = 0; i < 8; ++i) { r[i] = lo[i]; r[i + 8] = hi[i]; }
  return r;
}

// ---------------------------------------------------------------------------
// Pack fp32 row-major W[row0+k][n] (leading dim ld) into B-fragment layout f16.
// One 32-thread block per 32x16 (KxN) tile. Zero-pads k>=Krows or n>=Nvalid.
__global__ void pack_b_kernel(const float* __restrict__ W, int ld, int row0,
                              int Krows, int ntiles, int Nvalid,
                              _Float16* __restrict__ out) {
  const int tile = blockIdx.x;
  const int nt = tile % ntiles;
  const int kt = tile / ntiles;
  const int lane = threadIdx.x & 31;
  const int n = nt * 16 + (lane & 15);
  const int half = lane >> 4;
  v8h lo, hi;
#pragma unroll
  for (int i = 0; i < 16; ++i) {
    const int k = kt * 32 + (i < 8 ? half * 8 + i : 16 + half * 8 + (i - 8));  // lane's f16 i corresponds to K=half*8+i (i<8), 16+half*8+(i-8) (i>=8)
    float v = 0.0f;
    if (k < Krows && n < Nvalid) v = W[(size_t)(row0 + k) * ld + n];
    if (i < 8) lo[i] = (_Float16)v; else hi[i - 8] = (_Float16)v;
  }
  _Float16* dst = out + (size_t)tile * 512 + lane * 16;
  *(v8h*)dst = lo;
  *(v8h*)(dst + 8) = hi;
}

// ---------------------------------------------------------------------------
// WMMA GEMM: C[M,N] (fp32) = (accumulate ? C : 0) + A_f16[M,K] * Bpacked (+bias)
// One wave computes a 16-row x 64-col strip (4 N-tiles, A reused 4x).
// REQUIRES: ntiles % 4 == 0 and ktiles % 2 == 0 (guaranteed by caller padding),
// so the inner loop has NO divergent guards and is double-buffer pipelined.
__global__ void gemm_wmma_kernel(const _Float16* __restrict__ A, int lda,
                                 const _Float16* __restrict__ Bp,
                                 float* __restrict__ C, int ldc,
                                 const float* __restrict__ bias,
                                 int mtiles, int ntiles, int ktiles,
                                 int nvalid, int accumulate) {
  const int lane = threadIdx.x & 31;
  const int wid  = blockIdx.x * (blockDim.x >> 5) + (threadIdx.x >> 5);
  const int nblk = ntiles >> 2;
  if (wid >= mtiles * nblk) return;           // wave-uniform exit
  const int mt = wid / nblk;
  const int nb = wid % nblk;
  const int nt0 = nb * 4;
  const int n_lo = lane & 15;
  const int half = lane >> 4;

  v8f acc[4] = {};

  if (accumulate) {
#pragma unroll
    for (int j = 0; j < 4; ++j) {
      const int col = (nt0 + j) * 16 + n_lo;
      if (col < nvalid) {
#pragma unroll
        for (int r = 0; r < 8; ++r) {
          const int m = mt * 16 + r + half * 8;
          acc[j][r] = C[(size_t)m * ldc + col];
        }
      }
    }
  }

  // Double-buffered K pipeline (ktiles even).
  v16h aA = load_a_frag(A, lda, mt, 0, lane);
  v16h bA[4];
#pragma unroll
  for (int j = 0; j < 4; ++j) bA[j] = load_b_frag(Bp, ntiles, 0, nt0 + j, lane);

  for (int kt = 0; kt < ktiles; kt += 2) {
    // issue loads for kt+1 while WMMAs for kt execute
    v16h aB = load_a_frag(A, lda, mt, kt + 1, lane);
    v16h bB[4];
#pragma unroll
    for (int j = 0; j < 4; ++j) bB[j] = load_b_frag(Bp, ntiles, kt + 1, nt0 + j, lane);

#pragma unroll
    for (int j = 0; j < 4; ++j)
      acc[j] = __builtin_amdgcn_wmma_f32_16x16x32_f16(
          false, aA, false, bA[j], (short)0, acc[j], false, false);

    if (kt + 2 < ktiles) {  // wave-uniform scalar branch; refills buffer A in place
      aA = load_a_frag(A, lda, mt, kt + 2, lane);
#pragma unroll
      for (int j = 0; j < 4; ++j) bA[j] = load_b_frag(Bp, ntiles, kt + 2, nt0 + j, lane);
    }

#pragma unroll
    for (int j = 0; j < 4; ++j)
      acc[j] = __builtin_amdgcn_wmma_f32_16x16x32_f16(
          false, aB, false, bB[j], (short)0, acc[j], false, false);
  }

#pragma unroll
  for (int j = 0; j < 4; ++j) {
    const int col = (nt0 + j) * 16 + n_lo;
    if (col < nvalid) {
      const float bv = bias ? bias[col] : 0.0f;
#pragma unroll
      for (int r = 0; r < 8; ++r) {
        const int m = mt * 16 + r + half * 8;
        C[(size_t)m * ldc + col] = acc[j][r] + bv;
      }
    }
  }
}

// ---------------------------------------------------------------------------
// frames [B,Tenc,Din] fp32 -> A_f16 rows r = t*B + b, lda = Din
__global__ void frames_to_f16_kernel(const float* __restrict__ frames,
                                     _Float16* __restrict__ A) {
  const size_t total = (size_t)kTEnc * kB * kDIn;
  size_t idx = (size_t)blockIdx.x * blockDim.x + threadIdx.x;
  if (idx >= total) return;
  const int d = (int)(idx % kDIn);
  const int r = (int)(idx / kDIn);
  const int t = r / kB, b = r % kB;
  A[idx] = (_Float16)frames[((size_t)b * kTEnc + t) * kDIn + d];
}

// embeddings gather: A rows r = t*B + b, lda = 320 (cols >=300 zero)
__global__ void embed_to_f16_kernel(const float* __restrict__ emb,
                                    const int* __restrict__ caption,
                                    _Float16* __restrict__ A) {
  const size_t total = (size_t)kTDec * kB * kDWpad;
  size_t idx = (size_t)blockIdx.x * blockDim.x + threadIdx.x;
  if (idx >= total) return;
  const int d = (int)(idx % kDWpad);
  const int r = (int)(idx / kDWpad);
  const int t = r / kB, b = r % kB;
  float v = 0.0f;
  if (d < kDWord) {
    const int w = caption[b * kTDec + t];
    v = emb[(size_t)w * kDWord + d];
  }
  A[idx] = (_Float16)v;
}

__global__ void init_state_kernel(float* __restrict__ c, _Float16* __restrict__ h16) {
  const int idx = blockIdx.x * blockDim.x + threadIdx.x;
  if (idx >= kB * kH) return;
  c[idx] = 0.0f;
  h16[idx] = (_Float16)0.0f;
}

__device__ __forceinline__ float sigf(float x) { return 1.0f / (1.0f + __expf(-x)); }

// Z[b,4H] = x@Wx + h@Wh (already summed in Z); gates order i,j,f,o; forget_bias=1
__global__ void lstm_cell_kernel(const float* __restrict__ Z,
                                 const float* __restrict__ bias,
                                 float* __restrict__ c,
                                 _Float16* __restrict__ h16,
                                 _Float16* __restrict__ hs) {
  const int idx = blockIdx.x * blockDim.x + threadIdx.x;
  if (idx >= kB * kH) return;
  const int b = idx / kH, u = idx % kH;
  const float* zr = Z + (size_t)b * k4H;
  const float zi = zr[u]            + bias[u];
  const float zj = zr[kH + u]       + bias[kH + u];
  const float zf = zr[2 * kH + u]   + bias[2 * kH + u];
  const float zo = zr[3 * kH + u]   + bias[3 * kH + u];
  float cv = c[idx];
  cv = cv * sigf(zf + 1.0f) + sigf(zi) * tanhf(zj);
  const float h = tanhf(cv) * sigf(zo);
  c[idx] = cv;
  const _Float16 hh = (_Float16)h;
  h16[idx] = hh;
  if (hs) hs[idx] = hh;
}

// Cproj rows r = t*B + b (ld 320, bias already added) -> out[b, t, 0:300]
__global__ void write_out_kernel(const float* __restrict__ Cproj,
                                 float* __restrict__ out) {
  const size_t total = (size_t)kB * kTDec * kDWord;
  size_t idx = (size_t)blockIdx.x * blockDim.x + threadIdx.x;
  if (idx >= total) return;
  const int d = (int)(idx % kDWord);
  const int rest = (int)(idx / kDWord);
  const int t = rest % kTDec;
  const int b = rest / kTDec;
  out[idx] = Cproj[((size_t)t * kB + b) * kNPpad + d];
}

// ---------------------------------------------------------------------------
extern "C" void kernel_launch(void* const* d_in, const int* in_sizes, int n_in,
                              void* d_out, int out_size, void* d_ws, size_t ws_size,
                              hipStream_t stream) {
  (void)in_sizes; (void)n_in; (void)out_size; (void)ws_size;
  const float* frames  = (const float*)d_in[0];
  const float* emb     = (const float*)d_in[1];
  const float* W_enc   = (const float*)d_in[2];
  const float* b_enc   = (const float*)d_in[3];
  const float* W_dec   = (const float*)d_in[4];
  const float* b_dec   = (const float*)d_in[5];
  const float* W_proj  = (const float*)d_in[6];
  const float* b_proj  = (const float*)d_in[7];
  const int*   caption = (const int*)d_in[8];
  float* out = (float*)d_out;

  char* ws = (char*)d_ws;
  size_t off = 0;
  auto take = [&](size_t bytes) -> char* {
    char* p = ws + off;
    off = (off + bytes + 255) & ~(size_t)255;
    return p;
  };

  _Float16* WxE  = (_Float16*)take((size_t)kDIn  * k4H   * 2);  // packed enc x-weights
  _Float16* WhE  = (_Float16*)take((size_t)kH    * k4H   * 2);  // packed enc h-weights
  _Float16* WxD  = (_Float16*)take((size_t)kDWpad* k4H   * 2);  // packed dec x-weights
  _Float16* WhD  = (_Float16*)take((size_t)kH    * k4H   * 2);  // packed dec h-weights
  _Float16* WP   = (_Float16*)take((size_t)kH    * kNPpad* 2);  // packed proj weights
  _Float16* Aenc = (_Float16*)take((size_t)kTEnc * kB * kDIn * 2);
  float*    ZpE  = (float*)   take((size_t)kTEnc * kB * k4H  * 4);
  _Float16* Adec = (_Float16*)take((size_t)kTDec * kB * kDWpad * 2);
  float*    ZpD  = (float*)   take((size_t)kTDec * kB * k4H  * 4);
  _Float16* Hs   = (_Float16*)take((size_t)kTDec * kB * kH   * 2);
  _Float16* h16  = (_Float16*)take((size_t)kB * kH * 2);
  float*    cst  = (float*)   take((size_t)kB * kH * 4);
  float*    Cpj  = (float*)   take((size_t)kTDec * kB * kNPpad * 4);

  const int ntE = k4H / 16;                    // 128 N-tiles for 4H
  const int ntP = kNPpad / 16;                 // 20 N-tiles for proj
  // ---- pack weights into WMMA B-fragment layout (f16) ----
  pack_b_kernel<<<dim3((kDIn / 32) * ntE), 32, 0, stream>>>(W_enc, k4H, 0,     kDIn,  ntE, k4H,  WxE);
  pack_b_kernel<<<dim3((kH   / 32) * ntE), 32, 0, stream>>>(W_enc, k4H, kDIn,  kH,    ntE, k4H,  WhE);
  pack_b_kernel<<<dim3((kDWpad/32) * ntE), 32, 0, stream>>>(W_dec, k4H, 0,     kDWord,ntE, k4H,  WxD);
  pack_b_kernel<<<dim3((kH   / 32) * ntE), 32, 0, stream>>>(W_dec, k4H, kDWord,kH,    ntE, k4H,  WhD);
  pack_b_kernel<<<dim3((kH   / 32) * ntP), 32, 0, stream>>>(W_proj, kDWord, 0, kH, ntP, kDWord, WP);

  // ---- activations to f16 ----
  {
    const size_t tot = (size_t)kTEnc * kB * kDIn;
    frames_to_f16_kernel<<<dim3((unsigned)((tot + 255) / 256)), 256, 0, stream>>>(frames, Aenc);
  }
  {
    const size_t tot = (size_t)kTDec * kB * kDWpad;
    embed_to_f16_kernel<<<dim3((unsigned)((tot + 255) / 256)), 256, 0, stream>>>(emb, caption, Adec);
  }

  auto launch_gemm = [&](const _Float16* A, int lda, const _Float16* Bp, float* C,
                         int ldc, const float* bias, int mtiles, int ntiles,
                         int ktiles, int nvalid, int accum) {
    const int nblk = ntiles / 4;                  // ntiles % 4 == 0 guaranteed
    const int waves = mtiles * nblk;
    const int blocks = (waves + 7) / 8;           // 8 waves (256 threads) / block
    gemm_wmma_kernel<<<dim3(blocks), 256, 0, stream>>>(A, lda, Bp, C, ldc, bias,
                                                       mtiles, ntiles, ktiles,
                                                       nvalid, accum);
  };

  // ---- hoisted input GEMMs (all time steps at once) ----
  // ZpE[t*B+b, :] = frames_f16 @ Wx_enc
  launch_gemm(Aenc, kDIn, WxE, ZpE, k4H, nullptr,
              (kTEnc * kB) / 16, ntE, kDIn / 32, k4H, 0);
  // ZpD[t*B+b, :] = cap_emb_f16 @ Wx_dec
  launch_gemm(Adec, kDWpad, WxD, ZpD, k4H, nullptr,
              (kTDec * kB) / 16, ntE, kDWpad / 32, k4H, 0);

  init_state_kernel<<<dim3((kB * kH) / 256), 256, 0, stream>>>(cst, h16);

  // ---- encoder recurrence ----
  for (int t = 0; t < kTEnc; ++t) {
    float* Zt = ZpE + (size_t)t * kB * k4H;
    launch_gemm(h16, kH, WhE, Zt, k4H, nullptr, kB / 16, ntE, kH / 32, k4H, 1);
    lstm_cell_kernel<<<dim3((kB * kH) / 256), 256, 0, stream>>>(Zt, b_enc, cst, h16, nullptr);
  }

  // ---- decoder recurrence (state carried over from encoder) ----
  for (int t = 0; t < kTDec; ++t) {
    float* Zt = ZpD + (size_t)t * kB * k4H;
    launch_gemm(h16, kH, WhD, Zt, k4H, nullptr, kB / 16, ntE, kH / 32, k4H, 1);
    lstm_cell_kernel<<<dim3((kB * kH) / 256), 256, 0, stream>>>(
        Zt, b_dec, cst, h16, Hs + (size_t)t * kB * kH);
  }

  // ---- projection: Cpj = Hs @ W_proj + b_proj (N masked to 300) ----
  launch_gemm(Hs, kH, WP, Cpj, kNPpad, b_proj,
              (kTDec * kB) / 16, ntP, kH / 32, kDWord, 0);

  // ---- reorder [t,b] -> [b,t] into d_out ----
  {
    const size_t tot = (size_t)kB * kTDec * kDWord;
    write_out_kernel<<<dim3((unsigned)((tot + 255) / 256)), 256, 0, stream>>>(Cpj, out);
  }
}
